// ClusterMemory_8186207666552
// MI455X (gfx1250) — matrix operand order repaired and checked by reading, not compiled, not run on hardware
//
#include <hip/hip_runtime.h>

typedef __attribute__((ext_vector_type(16))) _Float16 v16h;
typedef __attribute__((ext_vector_type(8)))  float    v8f;

#define TEMP_INV  20.0f   // 1/0.05
#define LOGIT_MAX 20.0f   // |cos|<=1 -> logits in [-20,20]; fixed logsumexp shift
#define SUBT      50      // subtiles per chunk  (NT = 6250 = 125 * 50)
#define WAVES     8       // waves per block; block covers 8 row tiles = 128 rows

// ---------------- kernel 1: row L2-normalize (f32) ----------------
__global__ void k_normalize(const float* __restrict__ in, float* __restrict__ out, int D) {
  int b = blockIdx.x;
  int tid = threadIdx.x;                      // blockDim = 128 = D
  float x = in[(size_t)b * D + tid];
  float ss = x * x;
  #pragma unroll
  for (int m = 16; m >= 1; m >>= 1) ss += __shfl_xor(ss, m, 32);
  __shared__ float wred[4];
  if ((tid & 31) == 0) wred[tid >> 5] = ss;
  __syncthreads();
  float tot = wred[0] + wred[1] + wred[2] + wred[3];
  float inv = 1.0f / fmaxf(sqrtf(tot), 1e-12f);
  out[(size_t)b * D + tid] = x * inv;
}

// ------------- kernel 2: pack A fragments (f16, ISA 16-bit A layout) -------------
// fragment (rt,q): lane<16 -> row=lane,    halves = K[q*32+0..7],  K[q*32+16..23]
//                  lane>=16 -> row=lane-16, halves = K[q*32+8..15], K[q*32+24..31]
__global__ void k_packA(const float* __restrict__ inorm, _Float16* __restrict__ afrag) {
  int bi = blockIdx.x;                        // bi = rt*4 + q
  int rt = bi >> 2, q = bi & 3;
  int lane = threadIdx.x;                     // 32 threads
  int row = (rt << 4) + (lane & 15);
  int off = (lane < 16) ? 0 : 8;
  const float* src = inorm + (size_t)row * 128 + (q << 5) + off;
  _Float16* dst = afrag + (((size_t)bi) * 32 + lane) * 16;
  #pragma unroll
  for (int j = 0; j < 8; ++j) {
    dst[j]     = (_Float16)src[j];
    dst[8 + j] = (_Float16)src[16 + j];
  }
}

// ------------- kernel 3: pack B fragments (features -> f16, B layout) -------------
// fragment (t,q): lane<16 -> col n = t*16+lane,     halves = K[q*32+0..7],  K[q*32+16..23]
//                 lane>=16 -> col n = t*16+lane-16, halves = K[q*32+8..15], K[q*32+24..31]
__global__ void k_packB(const float* __restrict__ feat, _Float16* __restrict__ bfrag) {
  int bi = blockIdx.x;                        // bi = t*4 + q
  int lane = threadIdx.x;
  int t = bi >> 2, q = bi & 3;
  int n  = (t << 4) + (lane & 15);
  int k0 = (q << 5) + ((lane < 16) ? 0 : 8);
  const float* src = feat + (size_t)n * 128 + k0;
  _Float16* dst = bfrag + (((size_t)bi) * 32 + lane) * 16;
  #pragma unroll
  for (int j = 0; j < 8; ++j) { dst[j] = (_Float16)src[j]; dst[8 + j] = (_Float16)src[16 + j]; }
}

// ------------- kernel 4: picked target logits (exact f32) -------------
__global__ void k_picked(const float* __restrict__ inorm, const float* __restrict__ feat,
                         const long long* __restrict__ indexes,
                         const long long* __restrict__ labels,
                         float* __restrict__ picked, int D) {
  int b = blockIdx.x;
  int tid = threadIdx.x;                      // 128
  long long tgt = labels[indexes[b]];
  float p = inorm[(size_t)b * D + tid] * feat[(size_t)tgt * D + tid];
  #pragma unroll
  for (int m = 16; m >= 1; m >>= 1) p += __shfl_xor(p, m, 32);
  __shared__ float wred[4];
  if ((tid & 31) == 0) wred[tid >> 5] = p;
  __syncthreads();
  if (tid == 0) picked[b] = (wred[0] + wred[1] + wred[2] + wred[3]) * TEMP_INV;
}

// ------------- kernel 5: WMMA GEMM + shifted sum-exp partials -------------
__device__ __forceinline__ void load_bfrag(const _Float16* __restrict__ bp, v16h b[4]) {
  b[0] = *(const v16h*)(bp);
  b[1] = *(const v16h*)(bp + 512);
  b[2] = *(const v16h*)(bp + 1024);
  b[3] = *(const v16h*)(bp + 1536);
}

__device__ __forceinline__ void accum_tile(const v16h a[4], const v16h b[4], float s[8]) {
  v8f c = {};
  c = __builtin_amdgcn_wmma_f32_16x16x32_f16(false, a[0], false, b[0], (short)0, c, false, false);
  c = __builtin_amdgcn_wmma_f32_16x16x32_f16(false, a[1], false, b[1], (short)0, c, false, false);
  c = __builtin_amdgcn_wmma_f32_16x16x32_f16(false, a[2], false, b[2], (short)0, c, false, false);
  c = __builtin_amdgcn_wmma_f32_16x16x32_f16(false, a[3], false, b[3], (short)0, c, false, false);
  #pragma unroll
  for (int r = 0; r < 8; ++r)
    s[r] += __expf(c[r] * TEMP_INV - LOGIT_MAX);       // args <= 0: no overflow
}

// grid = RB * NCHUNKS blocks (RB = RT/8), 256 threads = 8 waves.
// Wave w owns row tile rt = blockRow*8 + w; ALL waves walk the same 50 subtiles
// of the chunk together, so B-fragment loads hit WGP$ after the first wave.
__global__ void k_gemm_sumexp(const _Float16* __restrict__ afrag,
                              const _Float16* __restrict__ bfrag,
                              float* __restrict__ partial,
                              int RB, int NCHUNKS) {
  int tid    = threadIdx.x;
  int lane   = tid & 31;
  int wave   = __builtin_amdgcn_readfirstlane(tid >> 5);   // scalar wave id
  int rtb    = blockIdx.x % RB;
  int chunk  = blockIdx.x / RB;
  int rt     = rtb * WAVES + wave;                          // scalar row tile
  int t0     = chunk * SUBT;                                // scalar subtile base

  const _Float16* ap = afrag + (size_t)rt * 2048 + (size_t)lane * 16;
  v16h a[4];
  a[0] = *(const v16h*)(ap);
  a[1] = *(const v16h*)(ap + 512);
  a[2] = *(const v16h*)(ap + 1024);
  a[3] = *(const v16h*)(ap + 1536);

  float s[8];
  #pragma unroll
  for (int r = 0; r < 8; ++r) s[r] = 0.0f;

  const _Float16* bbase = bfrag + (size_t)t0 * 2048;
  v16h b0[4], b1[4];
  load_bfrag(bbase + (size_t)lane * 16, b0);

  // ping-pong double buffer; SUBT is even, all branches scalar-uniform
  for (int i = 0; i < SUBT; i += 2) {
    const _Float16* bp = bbase + (size_t)(i + 1) * 2048 + (size_t)lane * 16;
    load_bfrag(bp, b1);                                    // next subtile in flight
    if (wave == 0 && i + 3 < SUBT)                         // scalar-gated prefetch
      __builtin_prefetch(bbase + (size_t)(i + 3) * 2048 + (size_t)lane * 64, 0, 3);
    accum_tile(a, b0, s);
    if (i + 2 < SUBT)
      load_bfrag(bbase + (size_t)(i + 2) * 2048 + (size_t)lane * 16, b0);
    accum_tile(a, b1, s);
  }

  // reduce 16 column-slots per half-wave: rows r (lanes 0-15), r+8 (lanes 16-31)
  #pragma unroll
  for (int r = 0; r < 8; ++r) {
    #pragma unroll
    for (int m = 1; m < 16; m <<= 1) s[r] += __shfl_xor(s[r], m, 16);
  }

  float* prow = partial + ((size_t)rt * NCHUNKS + chunk) * 16;
  if (lane == 0) {
    #pragma unroll
    for (int r = 0; r < 8; ++r) prow[r] = s[r];
  }
  if (lane == 16) {
    #pragma unroll
    for (int r = 0; r < 8; ++r) prow[8 + r] = s[r];
  }
}

// ------------- kernel 6: combine chunks, loss, mean -> scalar -------------
__global__ void k_final(const float* __restrict__ partial, const float* __restrict__ picked,
                        float* __restrict__ out, int B, int NCHUNKS) {
  int tid = threadIdx.x;                      // 256
  float acc = 0.0f;
  for (int b = tid; b < B; b += 256) {
    int rt = b >> 4, r = b & 15;
    const float* p = partial + ((size_t)rt * NCHUNKS) * 16 + r;
    float S = 0.0f;
    for (int c = 0; c < NCHUNKS; ++c) S += p[(size_t)c * 16];
    acc += (LOGIT_MAX + logf(S)) - picked[b];
  }
  __shared__ float red[256];
  red[tid] = acc;
  __syncthreads();
  #pragma unroll
  for (int off = 128; off >= 1; off >>= 1) {
    if (tid < off) red[tid] += red[tid + off];
    __syncthreads();
  }
  if (tid == 0) out[0] = red[0] / (float)B;
}

extern "C" void kernel_launch(void* const* d_in, const int* in_sizes, int n_in,
                              void* d_out, int out_size, void* d_ws, size_t ws_size,
                              hipStream_t stream) {
  const float*     inputs   = (const float*)d_in[0];      // [B, D] f32
  const long long* indexes  = (const long long*)d_in[1];  // [B] i64
  const long long* labels   = (const long long*)d_in[2];  // [N] i64
  const float*     features = (const float*)d_in[3];      // [N, D] f32 (row-normalized)

  const int B  = in_sizes[1];          // 2048
  const int D  = 128;                  // fixed by tiling (4 K-chunks of 32)
  const int N  = in_sizes[2];          // 100000
  const int RT = B / 16;               // 128 row tiles
  const int RB = RT / WAVES;           // 16 row blocks (128 rows each)
  const int NT = N / 16;               // 6250 column subtiles
  const int NCHUNKS = NT / SUBT;       // 125 chunks of 50 subtiles (exact)

  // workspace carve-up (~28.2 MB total)
  char* ws = (char*)d_ws;
  size_t off = 0;
  auto al = [](size_t x) { return (x + 255) & ~(size_t)255; };
  float*    inorm   = (float*)(ws + off);    off = al(off + (size_t)B * D * 4);
  _Float16* afrag   = (_Float16*)(ws + off); off = al(off + (size_t)B * D * 2);
  _Float16* bfrag   = (_Float16*)(ws + off); off = al(off + (size_t)N * D * 2);
  float*    picked  = (float*)(ws + off);    off = al(off + (size_t)B * 4);
  float*    partial = (float*)(ws + off);    off = al(off + (size_t)RT * NCHUNKS * 16 * 4);
  (void)ws_size; (void)n_in; (void)out_size;

  k_normalize  <<<B,            128, 0, stream>>>(inputs, inorm, D);
  k_packA      <<<RT * 4,        32, 0, stream>>>(inorm, afrag);
  k_packB      <<<NT * 4,        32, 0, stream>>>(features, bfrag);
  k_picked     <<<B,            128, 0, stream>>>(inorm, features, indexes, labels, picked, D);
  k_gemm_sumexp<<<RB * NCHUNKS, 256, 0, stream>>>(afrag, bfrag, partial, RB, NCHUNKS);
  k_final      <<<1,            256, 0, stream>>>(partial, picked, (float*)d_out, B, NCHUNKS);
}
